// SyntaxesAttention_26250840113806
// MI455X (gfx1250) — matrix operand order repaired and checked by reading, not compiled, not run on hardware
//
#include <hip/hip_runtime.h>
#include <hip/hip_bf16.h>

// ---------------------------------------------------------------------------
// Types for CDNA5 WMMA (wave32). v16bf = 8 VGPRs of packed bf16, v8f = 8 VGPRs.
// ---------------------------------------------------------------------------
typedef __attribute__((ext_vector_type(16))) __bf16 v16bf;
typedef __attribute__((ext_vector_type(8)))  __bf16 v8bf;
typedef __attribute__((ext_vector_type(8)))  float  v8f;

#define WMMA_BF16(a, b, c) \
  __builtin_amdgcn_wmma_f32_16x16x32_bf16(false, (a), false, (b), (short)0, (c), false, false)

// gfx1250 async global->LDS copy path (guarded; falls back to plain loads).
#if __has_builtin(__builtin_amdgcn_global_load_async_to_lds_b128) && \
    __has_builtin(__builtin_amdgcn_s_wait_asynccnt)
#define HAVE_ASYNC_LDS 1
// The builtin wants (int4 AS1*, int4 AS3*, imm offset, imm cpol) per the
// compiler diagnostic ("... vector_size(4*sizeof(int)) int __device__ *").
typedef int v4i_vs __attribute__((vector_size(16)));
typedef __attribute__((address_space(1))) v4i_vs* as1_v4i;
typedef __attribute__((address_space(3))) v4i_vs* as3_v4i;
static __device__ __forceinline__ as1_v4i to_as1(const void* p) {
  return (as1_v4i)(uintptr_t)p;   // global generic addresses map 1:1 to AS1
}
static __device__ __forceinline__ as3_v4i to_as3(void* p) {
  // Generic LDS addresses carry the LDS offset in the low 32 bits.
  return (as3_v4i)(unsigned int)(uintptr_t)p;
}
#else
#define HAVE_ASYNC_LDS 0
#endif

static __device__ __forceinline__ unsigned short f2bf(float f) {
  unsigned int u = __float_as_uint(f);
  u += 0x7FFFu + ((u >> 16) & 1u);   // round-to-nearest-even
  return (unsigned short)(u >> 16);
}

// A-matrix fragment (16x32 bf16, MxK): lane L holds row M=L%16.
// v16bf elems 0..7  = K[base .. base+7],   base = k0 + (L>=16 ? 16 : 0)
// v16bf elems 8..15 = K[base+8 .. base+15]
static __device__ __forceinline__ v16bf load_a_frag(const unsigned short* p) {
  v8bf lo = *reinterpret_cast<const v8bf*>(p);
  v8bf hi = *reinterpret_cast<const v8bf*>(p + 8);
  return __builtin_shufflevector(lo, hi, 0, 1, 2, 3, 4, 5, 6, 7,
                                          8, 9, 10, 11, 12, 13, 14, 15);
}

// ---------------------------------------------------------------------------
// Elementwise fp32 -> bf16 conversion
// ---------------------------------------------------------------------------
__global__ void f32_to_bf16_kernel(const float* __restrict__ in,
                                   unsigned short* __restrict__ out, int n) {
  int i = blockIdx.x * blockDim.x + threadIdx.x;
  if (i < n) out[i] = f2bf(in[i]);
}

// Gather the last C=128 rows per batch: Xc[(b*128+c), :] = Xb[(b*4096+3968+c), :]
__global__ void gather_ctx_kernel(const unsigned short* __restrict__ Xb,
                                  unsigned short* __restrict__ Xc) {
  int i = blockIdx.x * blockDim.x + threadIdx.x;   // 512*1024 elements
  if (i >= 512 * 1024) return;
  int row = i >> 10, col = i & 1023;
  int b = row >> 7, c = row & 127;
  Xc[i] = Xb[((size_t)(b * 4096 + 3968 + c) << 10) + col];
}

// Rearrange K (b,c,h,d)->(b,h,c,d) and V (b,c,h,d)->(b,h,d,c) so WMMA B-frag
// reads are contiguous 32B loads.
__global__ void kv_rearrange_kernel(const unsigned short* __restrict__ Kt,
                                    const unsigned short* __restrict__ Vtm,
                                    unsigned short* __restrict__ Kc,
                                    unsigned short* __restrict__ Vt) {
  int i = blockIdx.x * blockDim.x + threadIdx.x;   // 512*1024 elements
  if (i >= 512 * 1024) return;
  int row = i >> 10, hd = i & 1023;
  int b = row >> 7, c = row & 127;
  int h = hd >> 6, d = hd & 63;
  int bh = b * 16 + h;
  Kc[(size_t)bh * 8192 + c * 64 + d] = Kt[i];
  Vt[(size_t)bh * 8192 + d * 128 + c] = Vtm[i];
}

// ---------------------------------------------------------------------------
// C(M,N) = A(M,K) @ W(N,K)^T, bf16 in / fp32 accumulate.
// Block = 256 threads = 8 waves (2 in M x 4 in N); wave tile = 64x64
// (16 accumulators -> 16 WMMAs per 8KB of fragment traffic).
// Block tile = 128(M) x 256(N).
// ---------------------------------------------------------------------------
template <bool OUT_BF16>
__global__ __launch_bounds__(256) void gemm_bf16_wmma_kernel(
    const unsigned short* __restrict__ A, const unsigned short* __restrict__ W,
    void* __restrict__ Cout, int M, int N, int K) {
  const int tid  = threadIdx.x;
  const int wave = tid >> 5;
  const int lane = tid & 31;
  const int lr   = lane & 15;
  const int hf   = lane >> 4;
  const int wm   = wave & 1;
  const int wn   = wave >> 1;
  const int m0   = blockIdx.y * 128 + wm * 64;
  const int n0   = blockIdx.x * 256 + wn * 64;

  v8f zero = {};
  v8f acc[4][4];
#pragma unroll
  for (int mt = 0; mt < 4; mt++)
#pragma unroll
    for (int nt = 0; nt < 4; nt++) acc[mt][nt] = zero;

  const unsigned short* arow[4];
#pragma unroll
  for (int mt = 0; mt < 4; mt++)
    arow[mt] = A + (size_t)(m0 + mt * 16 + lr) * K + hf * 16;
  const unsigned short* brow[4];
#pragma unroll
  for (int nt = 0; nt < 4; nt++)
    brow[nt] = W + (size_t)(n0 + nt * 16 + lr) * K + hf * 16;

  for (int k0 = 0; k0 < K; k0 += 32) {
    v16bf af[4];
#pragma unroll
    for (int mt = 0; mt < 4; mt++) af[mt] = load_a_frag(arow[mt] + k0);
    v16bf bf_[4];
#pragma unroll
    for (int nt = 0; nt < 4; nt++)
      bf_[nt] = *reinterpret_cast<const v16bf*>(brow[nt] + k0);

    if (k0 + 32 < K) {   // gfx1250 global_prefetch_b8 on next K tile
      __builtin_prefetch(arow[0] + k0 + 32, 0, 0);
      __builtin_prefetch(brow[0] + k0 + 32, 0, 0);
    }

#pragma unroll
    for (int mt = 0; mt < 4; mt++)
#pragma unroll
      for (int nt = 0; nt < 4; nt++)
        acc[mt][nt] = WMMA_BF16(af[mt], bf_[nt], acc[mt][nt]);
  }

  // Epilogue: C/D layout -> row = j + hf*8, col = lr within each 16x16 tile.
#pragma unroll
  for (int mt = 0; mt < 4; mt++)
#pragma unroll
    for (int nt = 0; nt < 4; nt++)
#pragma unroll
      for (int j = 0; j < 8; j++) {
        int row = m0 + mt * 16 + j + hf * 8;
        int col = n0 + nt * 16 + lr;
        if (OUT_BF16)
          ((unsigned short*)Cout)[(size_t)row * N + col] = f2bf(acc[mt][nt][j]);
        else
          ((float*)Cout)[(size_t)row * N + col] = acc[mt][nt][j];
      }
}

// ---------------------------------------------------------------------------
// Windowed attention core. One wave handles 16 query rows x full 128-context
// for one (b,h). All 4 waves of a block share (b,h) (4 | 256 query tiles), so
// the K tile (128x64) and V tile (64x128) are staged ONCE per block into LDS
// -- via gfx1250 async global->LDS copies when available -- and all WMMA
// B-fragments are then ds_load_b128 reads.
// ---------------------------------------------------------------------------
__global__ __launch_bounds__(128) void attention_wmma_kernel(
    const unsigned short* __restrict__ Qb,   // (B,S,H) bf16
    const unsigned short* __restrict__ Kc,   // (B*nH, C, dh) bf16
    const unsigned short* __restrict__ Vt,   // (B*nH, dh, C) bf16
    const int* __restrict__ amask,           // (B,S) int32
    unsigned short* __restrict__ Ob) {       // (B,S,H) bf16
  __shared__ unsigned short kls[128 * 64];        // 16 KB K tile
  __shared__ unsigned short vls[64 * 128];        // 16 KB V tile (transposed)
  __shared__ unsigned short plds[4 * 16 * 128];   // 16 KB P (per wave)

  const int tid  = threadIdx.x;
  const int wave = tid >> 5;
  const int lane = tid & 31;
  const int lr   = lane & 15;
  const int hf   = lane >> 4;
  const int bh   = blockIdx.x >> 6;                    // shared by all 4 waves
  const int qt   = ((blockIdx.x & 63) << 2) + wave;    // 256 query tiles of 16
  const int b    = bh >> 4;
  const int h    = bh & 15;
  const int s0   = qt << 4;

  // ---- Stage K and V tiles for this (b,h) into LDS (block-cooperative) ----
  const unsigned short* kg = Kc + (size_t)bh * 8192;
  const unsigned short* vg = Vt + (size_t)bh * 8192;
#if HAVE_ASYNC_LDS
#pragma unroll
  for (int it = 0; it < 8; ++it) {
    int e = (it * 128 + tid) * 8;   // 8 bf16 = 16B per b128 transfer
    __builtin_amdgcn_global_load_async_to_lds_b128(to_as1(kg + e),
                                                   to_as3(kls + e), 0, 0);
    __builtin_amdgcn_global_load_async_to_lds_b128(to_as1(vg + e),
                                                   to_as3(vls + e), 0, 0);
  }
  __builtin_amdgcn_s_wait_asynccnt(0);
#else
#pragma unroll
  for (int it = 0; it < 8; ++it) {
    int e = (it * 128 + tid) * 8;
    *reinterpret_cast<uint4*>(kls + e) = *reinterpret_cast<const uint4*>(kg + e);
    *reinterpret_cast<uint4*>(vls + e) = *reinterpret_cast<const uint4*>(vg + e);
  }
#endif
  __syncthreads();

  v8f zero = {};

  // Q A-fragments: dh=64 -> two K-steps of 32.
  const unsigned short* qrow =
      Qb + ((size_t)(b * 4096 + s0 + lr) << 10) + h * 64 + hf * 16;
  v16bf aq0 = load_a_frag(qrow);
  v16bf aq1 = load_a_frag(qrow + 32);

  // scores = Q @ K^T  (8 context tiles of 16); K B-frags from LDS.
  v8f sc[8];
#pragma unroll
  for (int ct = 0; ct < 8; ct++) {
    const unsigned short* kr = kls + (ct * 16 + lr) * 64 + hf * 16;
    v16bf b0 = *reinterpret_cast<const v16bf*>(kr);
    v16bf b1 = *reinterpret_cast<const v16bf*>(kr + 32);
    v8f a = zero;
    a = WMMA_BF16(aq0, b0, a);
    a = WMMA_BF16(aq1, b1, a);
    sc[ct] = a;
  }

  // Scale + mask. Column c allowed iff pad && (c==0 || ctx_start+c <= s).
  int padv[8];
#pragma unroll
  for (int ct = 0; ct < 8; ct++)
    padv[ct] = amask[b * 4096 + 3968 + ct * 16 + lr];
#pragma unroll
  for (int ct = 0; ct < 8; ct++)
#pragma unroll
    for (int j = 0; j < 8; j++) {
      int c = ct * 16 + lr;
      int s = s0 + j + hf * 8;
      bool ok = (padv[ct] != 0) && ((c == 0) || (3968 + c <= s));
      float v = sc[ct][j] * 0.125f;   // 1/sqrt(64)
      sc[ct][j] = ok ? v : -1.0e30f;
    }

  // Row-wise softmax (row lives across 16 lanes of one half-wave) and
  // normalized-P store into LDS as bf16.
  unsigned short* pl = plds + wave * 2048;
#pragma unroll
  for (int j = 0; j < 8; j++) {
    float m = -1.0e30f;
#pragma unroll
    for (int ct = 0; ct < 8; ct++) m = fmaxf(m, sc[ct][j]);
    for (int d = 1; d < 16; d <<= 1) m = fmaxf(m, __shfl_xor(m, d, 32));
    float e[8];
    float sum = 0.0f;
#pragma unroll
    for (int ct = 0; ct < 8; ct++) {
      e[ct] = __expf(sc[ct][j] - m);
      sum += e[ct];
    }
    for (int d = 1; d < 16; d <<= 1) sum += __shfl_xor(sum, d, 32);
    float inv = 1.0f / sum;
    int row = j + hf * 8;
#pragma unroll
    for (int ct = 0; ct < 8; ct++)
      pl[row * 128 + ct * 16 + lr] = f2bf(e[ct] * inv);
  }
  __syncthreads();

  // out = P @ V : K = 128 context in 4 steps of 32; N = dh = 64 (4 tiles).
  // Both A-frags (P) and B-frags (V) come from LDS.
  v8f oacc[4];
#pragma unroll
  for (int nt = 0; nt < 4; nt++) oacc[nt] = zero;
#pragma unroll
  for (int kc = 0; kc < 128; kc += 32) {
    v16bf ap = load_a_frag(pl + lr * 128 + kc + hf * 16);
#pragma unroll
    for (int nt = 0; nt < 4; nt++) {
      const unsigned short* vr = vls + (nt * 16 + lr) * 128 + kc + hf * 16;
      v16bf vb = *reinterpret_cast<const v16bf*>(vr);
      oacc[nt] = WMMA_BF16(ap, vb, oacc[nt]);
    }
  }

#pragma unroll
  for (int nt = 0; nt < 4; nt++)
#pragma unroll
    for (int j = 0; j < 8; j++) {
      int s = s0 + j + hf * 8;
      Ob[((size_t)(b * 4096 + s) << 10) + h * 64 + nt * 16 + lr] =
          f2bf(oacc[nt][j]);
    }
}

// ---------------------------------------------------------------------------
// Host-side pipeline
// ---------------------------------------------------------------------------
extern "C" void kernel_launch(void* const* d_in, const int* in_sizes, int n_in,
                              void* d_out, int out_size, void* d_ws,
                              size_t ws_size, hipStream_t stream) {
  (void)in_sizes; (void)n_in; (void)out_size; (void)ws_size;

  const float* x     = (const float*)d_in[0];
  const int*   amask = (const int*)d_in[1];
  const float* Wq    = (const float*)d_in[2];
  const float* Wk    = (const float*)d_in[3];
  const float* Wv    = (const float*)d_in[4];
  const float* Wo    = (const float*)d_in[5];
  float* out = (float*)d_out;

  char* ws = (char*)d_ws;
  size_t off = 0;
  auto alloc = [&](size_t bytes) -> void* {
    void* p = ws + off;
    off += (bytes + 255) & ~(size_t)255;
    return p;
  };

  unsigned short* Xb  = (unsigned short*)alloc(16384ull * 1024 * 2);  // bf16 inputs
  unsigned short* Wqb = (unsigned short*)alloc(1024ull * 1024 * 2);
  unsigned short* Wkb = (unsigned short*)alloc(1024ull * 1024 * 2);
  unsigned short* Wvb = (unsigned short*)alloc(1024ull * 1024 * 2);
  unsigned short* Wob = (unsigned short*)alloc(1024ull * 1024 * 2);
  unsigned short* Qb  = (unsigned short*)alloc(16384ull * 1024 * 2);  // Q proj
  unsigned short* Xc  = (unsigned short*)alloc(512ull * 1024 * 2);    // ctx rows
  unsigned short* Kt  = (unsigned short*)alloc(512ull * 1024 * 2);    // K proj
  unsigned short* Vtm = (unsigned short*)alloc(512ull * 1024 * 2);    // V proj
  unsigned short* Kc  = (unsigned short*)alloc(512ull * 1024 * 2);    // (b,h,c,d)
  unsigned short* Vt  = (unsigned short*)alloc(512ull * 1024 * 2);    // (b,h,d,c)
  unsigned short* Ob  = (unsigned short*)alloc(16384ull * 1024 * 2);  // attn out

  // 1) fp32 -> bf16
  f32_to_bf16_kernel<<<(16777216 + 255) / 256, 256, 0, stream>>>(x, Xb, 16777216);
  f32_to_bf16_kernel<<<4096, 256, 0, stream>>>(Wq, Wqb, 1048576);
  f32_to_bf16_kernel<<<4096, 256, 0, stream>>>(Wk, Wkb, 1048576);
  f32_to_bf16_kernel<<<4096, 256, 0, stream>>>(Wv, Wvb, 1048576);
  f32_to_bf16_kernel<<<4096, 256, 0, stream>>>(Wo, Wob, 1048576);

  // 2) Q = X @ Wq^T   (16384x1024x1024), block tile 128x256
  gemm_bf16_wmma_kernel<true><<<dim3(4, 128), 256, 0, stream>>>(
      Xb, Wqb, Qb, 16384, 1024, 1024);

  // 3) K/V projections on context rows only (512x1024x1024)
  gather_ctx_kernel<<<2048, 256, 0, stream>>>(Xb, Xc);
  gemm_bf16_wmma_kernel<true><<<dim3(4, 4), 256, 0, stream>>>(
      Xc, Wkb, Kt, 512, 1024, 1024);
  gemm_bf16_wmma_kernel<true><<<dim3(4, 4), 256, 0, stream>>>(
      Xc, Wvb, Vtm, 512, 1024, 1024);
  kv_rearrange_kernel<<<2048, 256, 0, stream>>>(Kt, Vtm, Kc, Vt);

  // 4) attention core: 16384 wave-tiles / 4 waves per block
  attention_wmma_kernel<<<4096, 128, 0, stream>>>(Qb, Kc, Vt, amask, Ob);

  // 5) out = Attn @ Wo^T  -> fp32
  gemm_bf16_wmma_kernel<false><<<dim3(4, 128), 256, 0, stream>>>(
      Ob, Wob, out, 16384, 1024, 1024);
}
